// ChannelAttention_41334765257059
// MI455X (gfx1250) — matrix operand order repaired and checked by reading, not compiled, not run on hardware
//
#include <hip/hip_runtime.h>
#include <hip/hip_bf16.h>
#include <cstdint>

// ---- problem constants (match reference) ----
#define BATCH 4
#define CCH   256     // C
#define NPIX  16384   // H*W
#define OUTC  50      // O
#define OPAD  64      // O padded to WMMA multiple

// ---- WMMA vector types (wave32, gfx1250) ----
typedef __attribute__((ext_vector_type(16))) __bf16 v16bf;
typedef __attribute__((ext_vector_type(8)))  __bf16 v8bf;
typedef __attribute__((ext_vector_type(8)))  float  v8f;

static __device__ __forceinline__ v16bf cat8(v8bf lo, v8bf hi) {
  return __builtin_shufflevector(lo, hi, 0,1,2,3,4,5,6,7,8,9,10,11,12,13,14,15);
}

// A fragment 16x32 bf16 (MxK): lane L -> m = m0 + (L&15), half = L>>4.
// elems 0..7 : K = k0 + 8*half + e      (contiguous 16B)
// elems 8..15: K = k0 + 16 + 8*half + e (contiguous 16B)
static __device__ __forceinline__ v16bf load_A_frag(const __bf16* A, int lda, int m0, int k0) {
  const int lane = threadIdx.x & 31;
  const int m = m0 + (lane & 15);
  const int half = lane >> 4;
  const __bf16* p = A + m * lda + k0 + 8 * half;
  v8bf lo = *(const v8bf*)p;
  v8bf hi = *(const v8bf*)(p + 16);
  return cat8(lo, hi);
}

// B fragment 32x16 bf16 (KxN): lane L -> n = n0 + (L&15), half = L>>4; elems 0..7: K = k0 + 8*half + e, elems 8..15: K = k0 + 16 + 8*half + e.
static __device__ __forceinline__ v16bf load_B_frag(const __bf16* Bm, int ldb, int k0, int n0) {
  const int lane = threadIdx.x & 31;
  const int half = lane >> 4;
  const __bf16* p = Bm + (k0 + 8 * half) * ldb + n0 + (lane & 15);
  v16bf r;
#pragma unroll
  for (int e = 0; e < 8; ++e) {
    r[e] = p[e * ldb];
    r[8 + e] = p[(16 + e) * ldb];
  }
  return r;
}

static __device__ __forceinline__ v8f wmma_bf16(v16bf a, v16bf b, v8f c) {
  // v_wmma_f32_16x16x32_bf16: (neg_a, A, neg_b, B, c_mod, C, reuse_a, reuse_b)
  return __builtin_amdgcn_wmma_f32_16x16x32_bf16(false, a, false, b, (short)0, c, false, false);
}

static __device__ __forceinline__ v8f zero8() {
  v8f z = {0.f, 0.f, 0.f, 0.f, 0.f, 0.f, 0.f, 0.f};
  return z;
}

// ---- CDNA5 async global->LDS copy (ASYNCcnt-tracked), 16B per lane ----
// VDST holds the per-lane LDS byte offset; low 32 bits of a generic LDS
// pointer are exactly that offset (flat->LDS mapping truncates addr[31:0]).
static __device__ __forceinline__ void async_copy_b128(const void* gsrc, void* ldst) {
  const uint32_t l = (uint32_t)(uintptr_t)ldst;
  const uint64_t a = (uint64_t)(uintptr_t)gsrc;
  asm volatile("global_load_async_to_lds_b128 %0, %1, off" :: "v"(l), "v"(a) : "memory");
}
static __device__ __forceinline__ void async_wait0() {
  asm volatile("s_wait_asynccnt 0x0" ::: "memory");
}

// ------------------------------------------------------------------
// K1: F[b, 0:64, n] = Wf_pad[64,256] * x[b, :, n]   (bf16 out)
// ------------------------------------------------------------------
#define WF_LD 264   // 256 + 8 pad, row stride 528B (16B-aligned, bank-rotated)
#define XT_LD 136   // 128 + 8 pad, row stride 272B

__global__ __launch_bounds__(256) void k_computeF(const float* __restrict__ x,
                                                  const float* __restrict__ Wf,
                                                  __bf16* __restrict__ F) {
  __shared__ __align__(16) __bf16 sWf[OPAD * WF_LD];
  __shared__ __align__(16) __bf16 sX[32 * XT_LD];
  const int tid = threadIdx.x;
  const int b  = blockIdx.y;
  const int n0 = blockIdx.x * 128;

  // stage W_f once (rows >= OUTC zero-padded); needs fp32->bf16 convert
  for (int i = tid; i < OPAD * CCH; i += 256) {
    const int m = i >> 8, k = i & 255;
    const float v = (m < OUTC) ? Wf[m * CCH + k] : 0.0f;
    sWf[m * WF_LD + k] = (__bf16)v;
  }

  v8f acc[4];
#pragma unroll
  for (int t = 0; t < 4; ++t) acc[t] = zero8();

  const size_t xbase = ((size_t)b * CCH) * NPIX + n0;
  for (int k0 = 0; k0 < CCH; k0 += 32) {
    __syncthreads();
    // stage x[b][k0..k0+31][n0..n0+127] as bf16 (conversion -> VALU path)
    for (int i = tid; i < 32 * 32; i += 256) {
      const int r = i >> 5, c4 = i & 31;
      const float4 v = *(const float4*)&x[xbase + (size_t)(k0 + r) * NPIX + c4 * 4];
      __bf16* d = &sX[r * XT_LD + c4 * 4];
      d[0] = (__bf16)v.x; d[1] = (__bf16)v.y; d[2] = (__bf16)v.z; d[3] = (__bf16)v.w;
    }
    if (k0 + 32 < CCH)  // pull next tile toward L2 (global_prefetch_b8)
      __builtin_prefetch(&x[xbase + (size_t)(k0 + 32 + (tid >> 3)) * NPIX + (tid & 7) * 16], 0, 1);
    __syncthreads();

    const int w = tid >> 5;
    const v16bf bf = load_B_frag(sX, XT_LD, 0, w * 16);
#pragma unroll
    for (int t = 0; t < 4; ++t) {
      const v16bf af = load_A_frag(sWf, WF_LD, t * 16, k0);
      acc[t] = wmma_bf16(af, bf, acc[t]);
    }
  }

  const int w = tid >> 5, lane = tid & 31;
  const int half = lane >> 4, nc = lane & 15;
#pragma unroll
  for (int t = 0; t < 4; ++t)
#pragma unroll
    for (int r = 0; r < 8; ++r) {
      const int m = t * 16 + r + 8 * half;
      F[((size_t)b * OPAD + m) * NPIX + n0 + w * 16 + nc] = (__bf16)acc[t][r];
    }
}

// ------------------------------------------------------------------
// K2: S[b, 0:64, c] += sum_n F[b,o,n] * x[b,c,n]   (fp32, K-split atomics)
// ------------------------------------------------------------------
#define SF_LD 40   // 32 + 8 pad, row stride 80B (16B-aligned)

__global__ __launch_bounds__(256) void k_computeS(const float* __restrict__ x,
                                                  const __bf16* __restrict__ F,
                                                  float* __restrict__ S) {
  __shared__ __align__(16) __bf16 sF[OPAD * SF_LD];   // [64][32+8]
  __shared__ __align__(16) __bf16 sXT[32 * XT_LD];    // x transposed: [k][c]
  const int tid = threadIdx.x;
  const int b  = blockIdx.z;
  const int c0 = blockIdx.y * 128;
  const int kc = blockIdx.x;   // 32 chunks of 512 over NPIX

  v8f acc[4];
#pragma unroll
  for (int t = 0; t < 4; ++t) acc[t] = zero8();

  for (int kk = 0; kk < 16; ++kk) {
    const int kg = kc * 512 + kk * 32;
    __syncthreads();
    // F tile [64][32] bf16: pure byte copy -> async global->LDS, 1 chunk/thread
    {
      const int m = tid >> 2, ch = tid & 3;   // 64 rows x 4 x 16B
      async_copy_b128(&F[((size_t)b * OPAD + m) * NPIX + kg + ch * 8],
                      &sF[m * SF_LD + ch * 8]);
    }
    // x^T tile needs fp32->bf16 convert: VALU path with LDS transpose
    for (int i = tid; i < 128 * 32; i += 256) {
      const int c = i >> 5, k = i & 31;
      const float v = x[((size_t)b * CCH + c0 + c) * NPIX + kg + k];
      sXT[k * XT_LD + c] = (__bf16)v;
    }
    async_wait0();
    __syncthreads();

    const int w = tid >> 5;
    const v16bf bf = load_B_frag(sXT, XT_LD, 0, w * 16);
#pragma unroll
    for (int t = 0; t < 4; ++t) {
      const v16bf af = load_A_frag(sF, SF_LD, t * 16, 0);
      acc[t] = wmma_bf16(af, bf, acc[t]);
    }
  }

  const int w = tid >> 5, lane = tid & 31;
  const int half = lane >> 4, nc = lane & 15;
#pragma unroll
  for (int t = 0; t < 4; ++t)
#pragma unroll
    for (int r = 0; r < 8; ++r) {
      const int m = t * 16 + r + 8 * half;
      const int c = c0 + w * 16 + nc;
      atomicAdd(&S[((size_t)b * OPAD + m) * CCH + c], acc[t][r]);
    }
}

// ------------------------------------------------------------------
// K3a: row softmax over C; store transposed bf16 St[b][c][o]
// ------------------------------------------------------------------
__global__ __launch_bounds__(256) void k_softmax(const float* __restrict__ S,
                                                 __bf16* __restrict__ St) {
  __shared__ float red[256];
  const int o = blockIdx.x;   // < OUTC
  const int b = blockIdx.y;
  const int c = threadIdx.x;
  const float v = S[((size_t)b * OPAD + o) * CCH + c];
  red[c] = v; __syncthreads();
  for (int s = 128; s > 0; s >>= 1) {
    if (c < s) red[c] = fmaxf(red[c], red[c + s]);
    __syncthreads();
  }
  const float m = red[0]; __syncthreads();
  const float e = __expf(v - m);
  red[c] = e; __syncthreads();
  for (int s = 128; s > 0; s >>= 1) {
    if (c < s) red[c] += red[c + s];
    __syncthreads();
  }
  St[((size_t)b * CCH + c) * OPAD + o] = (__bf16)(e / red[0]);
}

// ------------------------------------------------------------------
// K3b: M[b][c][o] = sum_k W_beta[c][k] * St[b][k][o]   (tiny GEMM)
// ------------------------------------------------------------------
__global__ __launch_bounds__(64) void k_computeM(const float* __restrict__ Wb,
                                                 const __bf16* __restrict__ St,
                                                 __bf16* __restrict__ M) {
  const int o = threadIdx.x;     // 0..63
  const int c = blockIdx.x;      // 0..255
  const int b = blockIdx.y;
  float s = 0.f;
  for (int k = 0; k < CCH; ++k)  // Wb[c][k] is wave-uniform -> scalar broadcast
    s += Wb[c * CCH + k] * (float)St[((size_t)b * CCH + k) * OPAD + o];
  M[((size_t)b * CCH + c) * OPAD + o] = (__bf16)s;
}

// ------------------------------------------------------------------
// K4: out[b][c][n] = sum_o M[b][c][o] * F[b][o][n] + x[b][c][n]
// ------------------------------------------------------------------
#define K4_LD 80   // 64 + 16 pad, row stride 160B (16B-aligned for async b128)

__global__ __launch_bounds__(256) void k_computeOut(const float* __restrict__ x,
                                                    const __bf16* __restrict__ F,
                                                    const __bf16* __restrict__ M,
                                                    float* __restrict__ out) {
  __shared__ __align__(16) __bf16 sM[CCH * K4_LD];    // 40 KB
  __shared__ __align__(16) __bf16 sFt[OPAD * K4_LD];  // 10 KB
  const int tid = threadIdx.x;
  const int b  = blockIdx.y;
  const int n0 = blockIdx.x * 64;

  // both tiles are already bf16 -> async global->LDS byte copies (16B/lane)
  for (int i = tid; i < CCH * 8; i += 256) {          // sM: 256 rows x 8 chunks
    const int c = i >> 3, ch = i & 7;
    async_copy_b128(&M[((size_t)b * CCH + c) * OPAD + ch * 8],
                    &sM[c * K4_LD + ch * 8]);
  }
  for (int i = tid; i < OPAD * 8; i += 256) {         // sFt: 64 rows x 8 chunks
    const int o = i >> 3, ch = i & 7;
    async_copy_b128(&F[((size_t)b * OPAD + o) * NPIX + n0 + ch * 8],
                    &sFt[o * K4_LD + ch * 8]);
  }
  async_wait0();
  __syncthreads();

  const int w = tid >> 5;
  const int mg = w >> 2;            // 0..1 -> m rows [mg*128, mg*128+128)
  const int ncol = (w & 3) * 16;    // column sub-tile
  v8f acc[8];
#pragma unroll
  for (int t = 0; t < 8; ++t) acc[t] = zero8();

#pragma unroll
  for (int k0 = 0; k0 < OPAD; k0 += 32) {
    const v16bf bf = load_B_frag(sFt, K4_LD, k0, ncol);
#pragma unroll
    for (int t = 0; t < 8; ++t) {
      const v16bf af = load_A_frag(sM, K4_LD, mg * 128 + t * 16, k0);
      acc[t] = wmma_bf16(af, bf, acc[t]);
    }
  }

  const int lane = tid & 31, half = lane >> 4, nc = lane & 15;
#pragma unroll
  for (int t = 0; t < 8; ++t)
#pragma unroll
    for (int r = 0; r < 8; ++r) {
      const int c = mg * 128 + t * 16 + r + 8 * half;
      const size_t idx = ((size_t)b * CCH + c) * NPIX + n0 + ncol + nc;
      out[idx] = acc[t][r] + x[idx];   // residual epilogue
    }
}

// ------------------------------------------------------------------
__global__ void k_zero(uint32_t* __restrict__ p, int n) {
  const int i = blockIdx.x * blockDim.x + threadIdx.x;
  if (i < n) p[i] = 0u;
}

extern "C" void kernel_launch(void* const* d_in, const int* in_sizes, int n_in,
                              void* d_out, int out_size, void* d_ws, size_t ws_size,
                              hipStream_t stream) {
  const float* x  = (const float*)d_in[0];   // [B,C,H,W]
  const float* Wf = (const float*)d_in[1];   // [O,C]
  const float* Wb = (const float*)d_in[2];   // [C,C]
  float* out = (float*)d_out;                // [B,C,H,W]

  char* ws = (char*)d_ws;
  // workspace layout (~8.9 MB total)
  __bf16* F_ws  = (__bf16*)(ws);                               // 4*64*16384*2 = 8 MB
  float*  S_ws  = (float*) (ws + 8388608);                     // 4*64*256*4   = 256 KB
  __bf16* St_ws = (__bf16*)(ws + 8388608 + 262144);            // 4*256*64*2   = 128 KB
  __bf16* M_ws  = (__bf16*)(ws + 8388608 + 262144 + 131072);   // 4*256*64*2   = 128 KB

  const int zwords = (262144 + 131072) / 4;  // zero S (atomic accum) + St (pad cols)
  k_zero<<<(zwords + 255) / 256, 256, 0, stream>>>((uint32_t*)(ws + 8388608), zwords);

  k_computeF  <<<dim3(NPIX / 128, BATCH),          256, 0, stream>>>(x, Wf, F_ws);
  k_computeS  <<<dim3(32, CCH / 128, BATCH),       256, 0, stream>>>(x, F_ws, S_ws);
  k_softmax   <<<dim3(OUTC, BATCH),                256, 0, stream>>>(S_ws, St_ws);
  k_computeM  <<<dim3(CCH, BATCH),                  64, 0, stream>>>(Wb, St_ws, M_ws);
  k_computeOut<<<dim3(NPIX / 64, BATCH),           256, 0, stream>>>(x, F_ws, M_ws, out);
}
